// ConditionalAttentionLayer_24842090840248
// MI455X (gfx1250) — matrix operand order repaired and checked by reading, not compiled, not run on hardware
//
#include <hip/hip_runtime.h>
#include <hip/hip_bf16.h>

typedef _Float16 v16h __attribute__((ext_vector_type(16)));
typedef _Float16 v8h  __attribute__((ext_vector_type(8)));
typedef float    v8f  __attribute__((ext_vector_type(8)));

#define NN   4096
#define INSF 256
#define OUTS 64
#define MM   4
#define LEAK 0.2f

// workspace byte offsets
#define XH_OFF    ((size_t)0)                               // [N,INS] f16
#define WT_OFF    (XH_OFF + (size_t)NN*INSF*2)              // [M,OUTS,INS] f16 (W transposed)
#define HT_OFF    (WT_OFF + (size_t)MM*OUTS*INSF*2)         // [M,OUTS,N] f16 (h transposed)
#define GAMMA_OFF (HT_OFF + (size_t)MM*OUTS*NN*2)           // [M*OUTS] f32
#define BETA_OFF  (GAMMA_OFF + (size_t)MM*OUTS*4)           // [M*OUTS] f32
#define ESRC_OFF  (BETA_OFF + (size_t)MM*OUTS*4)            // [M,N] f32
#define EDST_OFF  (ESRC_OFF + (size_t)MM*NN*4)              // [M,N] f32
#define ADJB_OFF  (EDST_OFF + (size_t)MM*NN*4)              // [N,128] u32 bit-packed adj

// ---------------- prep: x -> f16, W -> transposed f16 ----------------
__global__ void cal_prep_kernel(const float* __restrict__ x, const float* __restrict__ W,
                                _Float16* __restrict__ xh, _Float16* __restrict__ WT) {
  int idx = blockIdx.x * blockDim.x + threadIdx.x;
  int stride = gridDim.x * blockDim.x;
  for (int i = idx; i < NN * INSF; i += stride) xh[i] = (_Float16)x[i];
  for (int i = idx; i < MM * OUTS * INSF; i += stride) {
    int m = i / (OUTS * INSF);
    int r = i % (OUTS * INSF);
    int o = r / INSF;
    int k = r % INSF;
    WT[i] = (_Float16)W[(size_t)(m * INSF + k) * OUTS + o];
  }
}

// ---------------- pack adjacency into bits via wave ballot ----------------
__global__ void cal_pack_kernel(const int* __restrict__ adj, unsigned int* __restrict__ adjbits) {
  int gid = blockIdx.x * blockDim.x + threadIdx.x;
  int stride = gridDim.x * blockDim.x;   // multiple of 32
  for (int i = gid; i < NN * NN; i += stride) {
    unsigned long long b = __ballot(adj[i] > 0);
    if ((threadIdx.x & 31) == 0) adjbits[i >> 5] = (unsigned int)b;
  }
}

// ---------------- conditioner: pooled -> gamma/beta ----------------
__global__ void cal_cond_kernel(const float* __restrict__ x, const float* __restrict__ Wc,
                                const float* __restrict__ bc,
                                float* __restrict__ gamma, float* __restrict__ beta) {
  __shared__ float pooled[INSF];
  int t = threadIdx.x; // 256 threads
  float s = 0.f;
  for (int n = 0; n < NN; ++n) s += x[(size_t)n * INSF + t];
  pooled[t] = s * (1.0f / NN);
  __syncthreads();
  float g0 = 0.f, g1 = 0.f;
  for (int i = 0; i < INSF; ++i) {
    float p = pooled[i];
    g0 += p * Wc[(size_t)i * (2 * MM * OUTS) + t];
    g1 += p * Wc[(size_t)i * (2 * MM * OUTS) + t + MM * OUTS];
  }
  gamma[t] = g0 + bc[t];
  beta[t]  = g1 + bc[t + MM * OUTS];
}

// ---------------- film: h = FiLM(x@W), e_src/e_dst, hT f16 ----------------
__global__ __launch_bounds__(128) void cal_film_kernel(
    const _Float16* __restrict__ xh, const _Float16* __restrict__ WT,
    const float* __restrict__ gamma, const float* __restrict__ beta,
    const float* __restrict__ a1, const float* __restrict__ a2,
    _Float16* __restrict__ hT, float* __restrict__ e_src, float* __restrict__ e_dst) {
  const int m    = threadIdx.x >> 5;   // wave = mechanism
  const int lane = threadIdx.x & 31;
  const int rr   = lane & 15;
  const int half = lane >> 4;
  const int rowbase = blockIdx.x * 16;

  v8f z = {0.f,0.f,0.f,0.f,0.f,0.f,0.f,0.f};
  v8f c[4];
#pragma unroll
  for (int t = 0; t < 4; ++t) c[t] = z;

  for (int kt = 0; kt < INSF; kt += 32) {
    // A fragment: lane<16 -> row rr, K {0..7,16..23}; lane>=16 -> K {8..15,24..31}
    const _Float16* ap = xh + (size_t)(rowbase + rr) * INSF + kt + half * 8;
    v8h alo = *(const v8h*)ap;
    v8h ahi = *(const v8h*)(ap + 16);
    v16h a;
#pragma unroll
    for (int i = 0; i < 8; ++i) { a[i] = alo[i]; a[8 + i] = ahi[i]; }
#pragma unroll
    for (int t = 0; t < 4; ++t) {
      const _Float16* bp = WT + (size_t)(m * OUTS + t * 16 + rr) * INSF + kt + half * 8;
      v8h blo = *(const v8h*)bp;
      v8h bhi = *(const v8h*)(bp + 16);
      v16h b;
#pragma unroll
      for (int i = 0; i < 8; ++i) { b[i] = blo[i]; b[8 + i] = bhi[i]; }
      c[t] = __builtin_amdgcn_wmma_f32_16x16x32_f16(false, a, false, b, (short)0, c[t], false, false);
    }
  }

  float es[8], ed[8];
#pragma unroll
  for (int r = 0; r < 8; ++r) { es[r] = 0.f; ed[r] = 0.f; }
#pragma unroll
  for (int t = 0; t < 4; ++t) {
    int o = t * 16 + rr;
    float g  = gamma[m * OUTS + o];
    float bt = beta[m * OUTS + o];
    float w1 = a1[m * OUTS + o];
    float w2 = a2[m * OUTS + o];
#pragma unroll
    for (int r = 0; r < 8; ++r) {
      int row = rowbase + half * 8 + r;
      float hv = g * c[t][r] + bt;
      hT[(size_t)(m * OUTS + o) * NN + row] = (_Float16)hv;
      es[r] += hv * w1;
      ed[r] += hv * w2;
    }
  }
#pragma unroll
  for (int mask = 1; mask < 16; mask <<= 1) {
#pragma unroll
    for (int r = 0; r < 8; ++r) {
      es[r] += __shfl_xor(es[r], mask, 16);
      ed[r] += __shfl_xor(ed[r], mask, 16);
    }
  }
  if (rr == 0) {
#pragma unroll
    for (int r = 0; r < 8; ++r) {
      int row = rowbase + half * 8 + r;
      e_src[m * NN + row] = es[r];
      e_dst[m * NN + row] = ed[r];
    }
  }
}

// ---------------- attention: max-only pass1, pipelined WMMA w@h, deferred normalize + ELU ----------------
__global__ __launch_bounds__(128) void cal_attn_kernel(
    const unsigned int* __restrict__ adjbits,
    const float* __restrict__ e_src, const float* __restrict__ e_dst,
    const _Float16* __restrict__ hT, float* __restrict__ out) {
  __shared__ unsigned int adjb[16 * 128];   // 16 rows x 4096 adjacency bits (8 KB)
  __shared__ float rmx[MM * 16];            // per-row max
  __shared__ float rsinv[MM * 16];          // per-row 1/sum (written in pass 2)
  const int tid = threadIdx.x;
  const int rowbase = blockIdx.x * 16;

  // stage packed adjacency tile via CDNA5 async global->LDS copies (ASYNCcnt path)
  {
    const unsigned int* src = adjbits + (size_t)rowbase * 128;
    unsigned int ldsbase = (unsigned int)(uintptr_t)(&adjb[0]);
    for (int i = tid; i < 512; i += 128) {            // 512 x 16B chunks = 8 KB
      unsigned int voff  = (unsigned int)i * 16u;
      unsigned int laddr = ldsbase + voff;
      asm volatile("global_load_async_to_lds_b128 %0, %1, %2"
                   :: "v"(laddr), "v"(voff), "s"(src) : "memory");
    }
    asm volatile("s_wait_asynccnt 0x0" ::: "memory");
  }
  __syncthreads();

  const int m    = tid >> 5;     // wave = mechanism
  const int lane = tid & 31;
  const int rr   = lane & 15;
  const int half = lane >> 4;
  const float* edm = e_dst + m * NN;

  // pass 1: per-row masked max only (no exp on this path)
  for (int r = 0; r < 16; ++r) {
    float esr = e_src[m * NN + rowbase + r];
    float mx = -3.0e38f;
    for (int jj = 0; jj < 128; ++jj) {
      unsigned int word = adjb[r * 128 + jj];
      float v = esr + edm[jj * 32 + lane];
      v = v > 0.f ? v : LEAK * v;
      v = ((word >> lane) & 1u) ? v : -1.0e9f;
      mx = fmaxf(mx, v);
    }
#pragma unroll
    for (int mask = 1; mask < 32; mask <<= 1) mx = fmaxf(mx, __shfl_xor(mx, mask, 32));
    if (lane == 0) rmx[m * 16 + r] = mx;
  }
  __syncthreads();

  const float esr = e_src[m * NN + rowbase + rr];
  const float mxr = rmx[m * 16 + rr];
  const int kbase = half * 8;   // A-fragment K base: lane<16 -> {0..7,16..23}, else {8..15,24..31}

  v8f z = {0.f,0.f,0.f,0.f,0.f,0.f,0.f,0.f};
  v8f c[4];
#pragma unroll
  for (int t = 0; t < 4; ++t) c[t] = z;
  float wsum = 0.f;

  // per-tile B base pointers (column o = t*16+rr of h^T, K offset half*8)
  const _Float16* bp0 = hT + (size_t)(m * OUTS + 0 * 16 + rr) * NN + half * 8;
  const _Float16* bp1 = hT + (size_t)(m * OUTS + 1 * 16 + rr) * NN + half * 8;
  const _Float16* bp2 = hT + (size_t)(m * OUTS + 2 * 16 + rr) * NN + half * 8;
  const _Float16* bp3 = hT + (size_t)(m * OUTS + 3 * 16 + rr) * NN + half * 8;

  auto step = [&](unsigned int wd, float4 q0, float4 q1, float4 q2, float4 q3,
                  v8h b0l, v8h b0h, v8h b1l, v8h b1h,
                  v8h b2l, v8h b2h, v8h b3l, v8h b3h) {
    float ev[16] = {q0.x,q0.y,q0.z,q0.w, q1.x,q1.y,q1.z,q1.w,
                    q2.x,q2.y,q2.z,q2.w, q3.x,q3.y,q3.z,q3.w};
    wd >>= kbase;
    v16h a;
#pragma unroll
    for (int i = 0; i < 8; ++i) {
      float v = esr + ev[i];
      v = v > 0.f ? v : LEAK * v;
      float w0 = ((wd >> i) & 1u) ? __expf(v - mxr) : 0.f;
      float v2 = esr + ev[8 + i];
      v2 = v2 > 0.f ? v2 : LEAK * v2;
      float w1 = ((wd >> (16 + i)) & 1u) ? __expf(v2 - mxr) : 0.f;
      wsum += w0 + w1;
      a[i] = (_Float16)w0;
      a[8 + i] = (_Float16)w1;
    }
    v16h b;
#pragma unroll
    for (int i = 0; i < 8; ++i) { b[i] = b0l[i]; b[8 + i] = b0h[i]; }
    c[0] = __builtin_amdgcn_wmma_f32_16x16x32_f16(false, a, false, b, (short)0, c[0], false, false);
#pragma unroll
    for (int i = 0; i < 8; ++i) { b[i] = b1l[i]; b[8 + i] = b1h[i]; }
    c[1] = __builtin_amdgcn_wmma_f32_16x16x32_f16(false, a, false, b, (short)0, c[1], false, false);
#pragma unroll
    for (int i = 0; i < 8; ++i) { b[i] = b2l[i]; b[8 + i] = b2h[i]; }
    c[2] = __builtin_amdgcn_wmma_f32_16x16x32_f16(false, a, false, b, (short)0, c[2], false, false);
#pragma unroll
    for (int i = 0; i < 8; ++i) { b[i] = b3l[i]; b[8 + i] = b3h[i]; }
    c[3] = __builtin_amdgcn_wmma_f32_16x16x32_f16(false, a, false, b, (short)0, c[3], false, false);
  };

  // prologue: load jt = 0
  unsigned int word = adjb[rr * 128 + 0];
  const float* edp = edm + kbase;
  float4 e0 = *(const float4*)(edp);
  float4 e1 = *(const float4*)(edp + 4);
  float4 e2 = *(const float4*)(edp + 16);
  float4 e3 = *(const float4*)(edp + 20);
  v8h b0l = *(const v8h*)(bp0),     b0h = *(const v8h*)(bp0 + 16);
  v8h b1l = *(const v8h*)(bp1),     b1h = *(const v8h*)(bp1 + 16);
  v8h b2l = *(const v8h*)(bp2),     b2h = *(const v8h*)(bp2 + 16);
  v8h b3l = *(const v8h*)(bp3),     b3h = *(const v8h*)(bp3 + 16);

  for (int jt = 0; jt < 127; ++jt) {
    // prefetch jt+1 while current tile computes
    int jn = (jt + 1) * 32;
    unsigned int nword = adjb[rr * 128 + jt + 1];
    const float* nedp = edm + jn + kbase;
    float4 f0 = *(const float4*)(nedp);
    float4 f1 = *(const float4*)(nedp + 4);
    float4 f2 = *(const float4*)(nedp + 16);
    float4 f3 = *(const float4*)(nedp + 20);
    v8h n0l = *(const v8h*)(bp0 + jn), n0h = *(const v8h*)(bp0 + jn + 16);
    v8h n1l = *(const v8h*)(bp1 + jn), n1h = *(const v8h*)(bp1 + jn + 16);
    v8h n2l = *(const v8h*)(bp2 + jn), n2h = *(const v8h*)(bp2 + jn + 16);
    v8h n3l = *(const v8h*)(bp3 + jn), n3h = *(const v8h*)(bp3 + jn + 16);

    step(word, e0, e1, e2, e3, b0l, b0h, b1l, b1h, b2l, b2h, b3l, b3h);

    word = nword; e0 = f0; e1 = f1; e2 = f2; e3 = f3;
    b0l = n0l; b0h = n0h; b1l = n1l; b1h = n1h;
    b2l = n2l; b2h = n2h; b3l = n3l; b3h = n3h;
  }
  step(word, e0, e1, e2, e3, b0l, b0h, b1l, b1h, b2l, b2h, b3l, b3h);  // jt = 127

  // combine the two per-lane partial sums of each row, publish 1/sum
  wsum += __shfl_xor(wsum, 16, 32);
  if (lane < 16) rsinv[m * 16 + rr] = 1.0f / fmaxf(wsum, 1e-30f);
  __syncthreads();

  // normalize + ELU + store [N, M*OUTS]
  float rinv[8];
#pragma unroll
  for (int r = 0; r < 8; ++r) rinv[r] = rsinv[m * 16 + half * 8 + r];
#pragma unroll
  for (int t = 0; t < 4; ++t) {
    int col = m * OUTS + t * 16 + rr;
#pragma unroll
    for (int r = 0; r < 8; ++r) {
      int row = rowbase + half * 8 + r;
      float v = c[t][r] * rinv[r];
      out[(size_t)row * (MM * OUTS) + col] = v > 0.f ? v : __expf(v) - 1.f;
    }
  }
}

extern "C" void kernel_launch(void* const* d_in, const int* in_sizes, int n_in,
                              void* d_out, int out_size, void* d_ws, size_t ws_size,
                              hipStream_t stream) {
  const float* x   = (const float*)d_in[0];
  const int*   adj = (const int*)d_in[1];
  const float* W   = (const float*)d_in[2];
  const float* a1  = (const float*)d_in[3];
  const float* a2  = (const float*)d_in[4];
  const float* Wc  = (const float*)d_in[5];
  const float* bc  = (const float*)d_in[6];
  float* out = (float*)d_out;
  char* ws = (char*)d_ws;

  _Float16* xh = (_Float16*)(ws + XH_OFF);
  _Float16* WT = (_Float16*)(ws + WT_OFF);
  _Float16* hT = (_Float16*)(ws + HT_OFF);
  float* gamma = (float*)(ws + GAMMA_OFF);
  float* beta  = (float*)(ws + BETA_OFF);
  float* esrc  = (float*)(ws + ESRC_OFF);
  float* edst  = (float*)(ws + EDST_OFF);
  unsigned int* adjb = (unsigned int*)(ws + ADJB_OFF);

  cal_prep_kernel<<<1024, 256, 0, stream>>>(x, W, xh, WT);
  cal_pack_kernel<<<4096, 256, 0, stream>>>(adj, adjb);
  cal_cond_kernel<<<1, 256, 0, stream>>>(x, Wc, bc, gamma, beta);
  cal_film_kernel<<<NN / 16, 128, 0, stream>>>(xh, WT, gamma, beta, a1, a2, hT, esrc, edst);
  cal_attn_kernel<<<NN / 16, 128, 0, stream>>>(adjb, esrc, edst, hT, out);
}
